// EdgeAggregation_39058432590468
// MI455X (gfx1250) — matrix operand order repaired and checked by reading, not compiled, not run on hardware
//
#include <hip/hip_runtime.h>
#include <hip/hip_bf16.h>

typedef __attribute__((ext_vector_type(16))) _Float16 v16h;
typedef __attribute__((ext_vector_type(8)))  _Float16 v8h;
typedef __attribute__((ext_vector_type(8)))  float    v8f;
typedef __attribute__((ext_vector_type(4)))  float    v4f;

#define D_DIM 128
#define H_DIM 256
#define E_NUM 600000
#define NTILES (E_NUM / 16)

#define W2_PITCH 264      // halves per row (256 + 8 pad) -> 528B rows, 16B aligned
#define WO_PITCH 136      // halves per row (128 + 8 pad) -> 272B rows, 16B aligned
#define SCR_PITCH 264
#define W2_BYTES (256 * W2_PITCH * 2)              // 135168
#define WO_BYTES (256 * WO_PITCH * 2)              // 69632
#define SCR_BYTES_PER_WAVE (16 * SCR_PITCH * 2)    // 8448
#define NWAVES 8
#define LDS_TOTAL (W2_BYTES + WO_BYTES + NWAVES * SCR_BYTES_PER_WAVE + NWAVES * 32 * 4) // 273408

// d_ws layout (bytes)
#define WS_W1H 0          // 128*256 f16 = 65536
#define WS_W2H 65536      // 256*256 f16 = 131072
#define WS_WOH 196608     // 256*128 f16 = 65536
#define WS_B1P 262144     // 256 f32
#define WS_B2P 263168     // 256 f32

static __device__ __forceinline__ v8f wmma_f16(v16h a, v16h b, v8f c) {
  return __builtin_amdgcn_wmma_f32_16x16x32_f16(false, a, false, b, (short)0, c, false, false);
}

// ---- prep kernels: fold LN affine params into f16 weights / f32 biases ----
extern "C" __global__ void prep_fold_w1(const float* __restrict__ w1, const float* __restrict__ g,
                                        _Float16* __restrict__ w1h) {
  int i = blockIdx.x * 256 + threadIdx.x;
  if (i < D_DIM * H_DIM) w1h[i] = (_Float16)(w1[i] * g[i >> 8]);
}
extern "C" __global__ void prep_fold_w2(const float* __restrict__ w2, const float* __restrict__ g,
                                        _Float16* __restrict__ w2h) {
  int i = blockIdx.x * 256 + threadIdx.x;
  if (i < H_DIM * H_DIM) w2h[i] = (_Float16)(w2[i] * g[i >> 8]);
}
extern "C" __global__ void prep_conv_wo(const float* __restrict__ wo, _Float16* __restrict__ woh) {
  int i = blockIdx.x * 256 + threadIdx.x;
  if (i < H_DIM * D_DIM) woh[i] = (_Float16)wo[i];
}
extern "C" __global__ void prep_bias(const float* __restrict__ b1, const float* __restrict__ ln1_b,
                                     const float* __restrict__ w1,
                                     const float* __restrict__ b2, const float* __restrict__ ln2_b,
                                     const float* __restrict__ w2,
                                     float* __restrict__ b1p, float* __restrict__ b2p) {
  int n = threadIdx.x;
  if (blockIdx.x == 0) {
    float acc = b1[n];
    for (int k = 0; k < D_DIM; ++k) acc += ln1_b[k] * w1[k * H_DIM + n];
    b1p[n] = acc;
  } else {
    float acc = b2[n];
    for (int k = 0; k < H_DIM; ++k) acc += ln2_b[k] * w2[k * H_DIM + n];
    b2p[n] = acc;
  }
}

extern "C" __global__ void __launch_bounds__(256, 1)
edge_mlp_scatter_kernel(const float* __restrict__ edges,
                        const int* __restrict__ eidx,
                        const _Float16* __restrict__ w1h,
                        const _Float16* __restrict__ w2h,
                        const _Float16* __restrict__ woh,
                        const float* __restrict__ b1p,
                        const float* __restrict__ b2p,
                        const float* __restrict__ wg,
                        const float* __restrict__ bg,
                        const float* __restrict__ bo,
                        float* __restrict__ out) {
  extern __shared__ char smem[];
  _Float16* w2L = (_Float16*)smem;
  _Float16* woL = (_Float16*)(smem + W2_BYTES);

  const int tid  = threadIdx.x;
  const int lane = tid & 31;
  const int wv   = tid >> 5;
  const int hi16 = lane >> 4;     // 0: lanes 0-15, 1: lanes 16-31
  const int nsub = lane & 15;

  _Float16* scr = (_Float16*)(smem + W2_BYTES + WO_BYTES + wv * SCR_BYTES_PER_WAVE);
  float* stats  = (float*)(smem + W2_BYTES + WO_BYTES + NWAVES * SCR_BYTES_PER_WAVE) + wv * 32;

  // ---- stage folded f16 w2 / wo into padded LDS ----
  for (int i = tid; i < 256 * 32; i += 256) {
    int row = i >> 5, c = (i & 31) * 8;
    *(v8h*)(w2L + row * W2_PITCH + c) = *(const v8h*)(w2h + row * 256 + c);
  }
  for (int i = tid; i < 256 * 16; i += 256) {
    int row = i >> 4, c = (i & 15) * 8;
    *(v8h*)(woL + row * WO_PITCH + c) = *(const v8h*)(woh + row * 128 + c);
  }
  __syncthreads();

  // ---- hoist per-lane loop-invariant vectors ----
  float b1v[16], b2v[16], wgv[16], bov[8];
  #pragma unroll
  for (int t = 0; t < 16; ++t) {
    b1v[t] = b1p[t * 16 + nsub];
    b2v[t] = b2p[t * 16 + nsub];
    wgv[t] = wg[t * 16 + nsub];
  }
  #pragma unroll
  for (int t = 0; t < 8; ++t) bov[t] = bo[t * 16 + nsub];
  const float bgv = bg[0];

  const int waveGlobal = blockIdx.x * NWAVES + wv;
  const int waveCount  = gridDim.x * NWAVES;

  for (int tile = waveGlobal; tile < NTILES; tile += waveCount) {
    const int e0 = tile * 16;

    // ---- prefetch next tile's edge rows + indices (hides HBM latency under GEMMs) ----
    const int tileNext = tile + waveCount;
    if (tileNext < NTILES) {                      // wave-uniform guard
      const char* p = (const char*)(edges + (size_t)(tileNext * 16 + nsub) * D_DIM) + hi16 * 256;
      __builtin_prefetch(p, 0, 1);
      __builtin_prefetch(p + 128, 0, 1);
      __builtin_prefetch(eidx + tileNext * 16, 0, 1);
    }

    // ---- edge pass: f32 LN1 stats + stash raw f16 rows into LDS scratch ----
    const float* erow = edges + (size_t)(e0 + nsub) * D_DIM;
    float s = 0.f, q = 0.f;
    #pragma unroll
    for (int j = 0; j < 8; ++j) {
      int blk = 2 * j + hi16;
      v4f a = *(const v4f*)(erow + blk * 8);
      v4f b = *(const v4f*)(erow + blk * 8 + 4);
      #pragma unroll
      for (int u = 0; u < 4; ++u) { s += a[u] + b[u]; q += a[u] * a[u] + b[u] * b[u]; }
      v8h h;
      h[0]=(_Float16)a[0]; h[1]=(_Float16)a[1]; h[2]=(_Float16)a[2]; h[3]=(_Float16)a[3];
      h[4]=(_Float16)b[0]; h[5]=(_Float16)b[1]; h[6]=(_Float16)b[2]; h[7]=(_Float16)b[3];
      *(v8h*)(scr + nsub * SCR_PITCH + blk * 8) = h;
    }
    s += __shfl_xor(s, 16); q += __shfl_xor(q, 16);
    float mu   = s * (1.f / 128.f);
    float rstd = rsqrtf(q * (1.f / 128.f) - mu * mu + 1e-5f);
    asm volatile("s_wait_dscnt 0" ::: "memory");

    // ---- A1 fragments: packed-f16 normalize (LN gamma/beta folded into w1h/b1p) ----
    const _Float16 mu1h = (_Float16)mu, rs1h = (_Float16)rstd;
    v16h A1[4];
    #pragma unroll
    for (int kb = 0; kb < 4; ++kb) {
      union { v16h v; v8h h2[2]; } fa;
      #pragma unroll
      for (int part = 0; part < 2; ++part) {
        int k0 = kb * 32 + part * 16 + 8 * hi16;
        v8h hv = *(const v8h*)(scr + nsub * SCR_PITCH + k0);
        fa.h2[part] = (hv - mu1h) * rs1h;
      }
      A1[kb] = fa.v;
    }

    // ---- GEMM1: [16x128] x w1'[128x256]; ELU; stash h1; LN2 stats ----
    float sA[8], qA[8];
    #pragma unroll
    for (int v = 0; v < 8; ++v) { sA[v] = 0.f; qA[v] = 0.f; }
    for (int t = 0; t < 16; ++t) {
      v16h Bf[4];
      #pragma unroll
      for (int kb = 0; kb < 4; ++kb) {
        const _Float16* bp = w1h + (size_t)(kb * 32 + 8 * hi16) * H_DIM + t * 16 + nsub;
        v16h fb;
        #pragma unroll
        for (int i = 0; i < 8; ++i) { fb[i] = bp[i * H_DIM]; fb[8 + i] = bp[(16 + i) * H_DIM]; }
        Bf[kb] = fb;
      }
      v8f acc0 = {}, acc1 = {};
      acc0 = wmma_f16(A1[0], Bf[0], acc0);
      acc1 = wmma_f16(A1[1], Bf[1], acc1);
      acc0 = wmma_f16(A1[2], Bf[2], acc0);
      acc1 = wmma_f16(A1[3], Bf[3], acc1);
      const float bb = b1v[t];
      const int n = t * 16 + nsub;
      #pragma unroll
      for (int v = 0; v < 8; ++v) {
        float x = (acc0[v] + acc1[v]) + bb;
        x = x > 0.f ? x : (__expf(x) - 1.f);          // ELU
        sA[v] += x; qA[v] += x * x;
        scr[(v + 8 * hi16) * SCR_PITCH + n] = (_Float16)x;
      }
    }

    // ---- LN2 stats: butterfly within 16-lane group; publish via LDS ----
    #pragma unroll
    for (int v = 0; v < 8; ++v) {
      #pragma unroll
      for (int off = 1; off < 16; off <<= 1) {
        sA[v] += __shfl_xor(sA[v], off);
        qA[v] += __shfl_xor(qA[v], off);
      }
    }
    if (lane == 0 || lane == 16) {
      #pragma unroll
      for (int v = 0; v < 8; ++v) {
        float m2 = sA[v] * (1.f / 256.f);
        float vv = qA[v] * (1.f / 256.f) - m2 * m2;
        stats[8 * hi16 + v]      = m2;
        stats[16 + 8 * hi16 + v] = rsqrtf(vv + 1e-5f);
      }
    }
    asm volatile("s_wait_dscnt 0" ::: "memory");
    const _Float16 mu2h = (_Float16)stats[nsub];
    const _Float16 rs2h = (_Float16)stats[16 + nsub];

    // ---- A2 fragments: packed-f16 normalize (gamma/beta folded into w2h/b2p) ----
    v16h A2[8];
    #pragma unroll
    for (int kb = 0; kb < 8; ++kb) {
      union { v16h v; v8h h2[2]; } fa;
      #pragma unroll
      for (int part = 0; part < 2; ++part) {
        int k0 = kb * 32 + part * 16 + 8 * hi16;
        v8h hv = *(const v8h*)(scr + nsub * SCR_PITCH + k0);
        fa.h2[part] = (hv - mu2h) * rs2h;
      }
      A2[kb] = fa.v;
    }

    // ---- GEMM2: [16x256] x w2'[256x256]; gate partials; stash h3 ----
    float gp[8];
    #pragma unroll
    for (int v = 0; v < 8; ++v) gp[v] = 0.f;
    for (int t = 0; t < 16; ++t) {
      v16h Bf[8];
      #pragma unroll
      for (int kb = 0; kb < 8; ++kb) {
        const _Float16* bp = w2L + (kb * 32 + 8 * hi16) * W2_PITCH + t * 16 + nsub;
        v16h fb;
        #pragma unroll
        for (int i = 0; i < 8; ++i) { fb[i] = bp[i * W2_PITCH]; fb[8 + i] = bp[(16 + i) * W2_PITCH]; }
        Bf[kb] = fb;
      }
      v8f acc0 = {}, acc1 = {};
      #pragma unroll
      for (int kb = 0; kb < 8; kb += 2) {
        acc0 = wmma_f16(A2[kb],     Bf[kb],     acc0);
        acc1 = wmma_f16(A2[kb + 1], Bf[kb + 1], acc1);
      }
      const float bb = b2v[t], wgn = wgv[t];
      const int n = t * 16 + nsub;
      #pragma unroll
      for (int v = 0; v < 8; ++v) {
        float x = (acc0[v] + acc1[v]) + bb;
        gp[v] += x * wgn;
        scr[(v + 8 * hi16) * SCR_PITCH + n] = (_Float16)x;
      }
    }

    // ---- gate = sigmoid(h3 @ wg + bg) per row ----
    #pragma unroll
    for (int v = 0; v < 8; ++v) {
      #pragma unroll
      for (int off = 1; off < 16; off <<= 1) gp[v] += __shfl_xor(gp[v], off);
      gp[v] = 1.f / (1.f + __expf(-(gp[v] + bgv)));
    }

    // ---- A3 fragments: raw f16 copy of h3 ----
    asm volatile("s_wait_dscnt 0" ::: "memory");
    v16h A3[8];
    #pragma unroll
    for (int kb = 0; kb < 8; ++kb) {
      union { v16h v; v8h h2[2]; } fa;
      int k0 = kb * 32 + 8 * hi16;
      fa.h2[0] = *(const v8h*)(scr + nsub * SCR_PITCH + k0);
      fa.h2[1] = *(const v8h*)(scr + nsub * SCR_PITCH + k0 + 16);
      A3[kb] = fa.v;
    }

    int nodeIdx[8];
    #pragma unroll
    for (int v = 0; v < 8; ++v) nodeIdx[v] = eidx[e0 + v + 8 * hi16];

    // ---- GEMM3: [16x256] x wo[256x128]; gate; atomic scatter-add ----
    for (int t = 0; t < 8; ++t) {
      v16h Bf[8];
      #pragma unroll
      for (int kb = 0; kb < 8; ++kb) {
        const _Float16* bp = woL + (kb * 32 + 8 * hi16) * WO_PITCH + t * 16 + nsub;
        v16h fb;
        #pragma unroll
        for (int i = 0; i < 8; ++i) { fb[i] = bp[i * WO_PITCH]; fb[8 + i] = bp[(16 + i) * WO_PITCH]; }
        Bf[kb] = fb;
      }
      v8f acc0 = {}, acc1 = {};
      #pragma unroll
      for (int kb = 0; kb < 8; kb += 2) {
        acc0 = wmma_f16(A3[kb],     Bf[kb],     acc0);
        acc1 = wmma_f16(A3[kb + 1], Bf[kb + 1], acc1);
      }
      const float bon = bov[t];
      const int n = t * 16 + nsub;
      #pragma unroll
      for (int v = 0; v < 8; ++v) {
        float x = ((acc0[v] + acc1[v]) + bon) * gp[v];
        atomicAdd(out + (size_t)nodeIdx[v] * D_DIM + n, x);
      }
    }
  }
}

extern "C" void kernel_launch(void* const* d_in, const int* in_sizes, int n_in,
                              void* d_out, int out_size, void* d_ws, size_t ws_size,
                              hipStream_t stream) {
  const float* edges = (const float*)d_in[0];
  const int*   eidx  = (const int*)d_in[1];
  const float* ln1_g = (const float*)d_in[3];
  const float* ln1_b = (const float*)d_in[4];
  const float* w1    = (const float*)d_in[5];
  const float* b1    = (const float*)d_in[6];
  const float* ln2_g = (const float*)d_in[7];
  const float* ln2_b = (const float*)d_in[8];
  const float* w2    = (const float*)d_in[9];
  const float* b2    = (const float*)d_in[10];
  const float* wg    = (const float*)d_in[11];
  const float* bg    = (const float*)d_in[12];
  const float* wo    = (const float*)d_in[13];
  const float* bo    = (const float*)d_in[14];
  float* out = (float*)d_out;

  char* ws = (char*)d_ws;
  _Float16* w1h = (_Float16*)(ws + WS_W1H);
  _Float16* w2h = (_Float16*)(ws + WS_W2H);
  _Float16* woh = (_Float16*)(ws + WS_WOH);
  float*    b1p = (float*)(ws + WS_B1P);
  float*    b2p = (float*)(ws + WS_B2P);

  hipMemsetAsync(d_out, 0, (size_t)out_size * sizeof(float), stream);
  prep_fold_w1<<<(D_DIM * H_DIM + 255) / 256, 256, 0, stream>>>(w1, ln1_g, w1h);
  prep_fold_w2<<<(H_DIM * H_DIM + 255) / 256, 256, 0, stream>>>(w2, ln2_g, w2h);
  prep_conv_wo<<<(H_DIM * D_DIM + 255) / 256, 256, 0, stream>>>(wo, woh);
  prep_bias<<<2, 256, 0, stream>>>(b1, ln1_b, w1, b2, ln2_b, w2, b1p, b2p);

  (void)hipFuncSetAttribute((const void*)edge_mlp_scatter_kernel,
                            hipFuncAttributeMaxDynamicSharedMemorySize, LDS_TOTAL);
  edge_mlp_scatter_kernel<<<dim3(1024), dim3(256), LDS_TOTAL, stream>>>(
      edges, eidx, w1h, w2h, woh, b1p, b2p, wg, bg, bo, out);
}
